// MultiHeadAttention_15942918603425
// MI455X (gfx1250) — matrix operand order repaired and checked by reading, not compiled, not run on hardware
//
#include <hip/hip_runtime.h>
#include <hip/hip_bf16.h>

// ---------------------------------------------------------------------------
// MI455X (gfx1250) fused MHA: bf16 WMMA everywhere, flash-softmax attention,
// software-pipelined with a one-iteration shift (fetch i+2 / stage i+1 /
// compute i) so global loads stay in flight across a barrier + full compute.
//   B=2 Q=2048 K=2048 C=512 H=8 D=64  (H*D == C == 512)
// Workspace layout (bytes):
//   [0,4M)    Qp  bf16 [B,H,Q,D]  (pre-scaled by 1/sqrt(D))
//   [4M,8M)   Kp  bf16 [B,H,K,D]
//   [8M,12M)  Vp  bf16 [B,H,K,D]
//   [12M,20M) G   f32  [B*Q, H*D] (sigmoid gate)
//   [20M,24M) Ows bf16 [B*Q, H*D] (gated attention output)
// ---------------------------------------------------------------------------

typedef __attribute__((ext_vector_type(16))) __bf16 v16bf;
typedef __attribute__((ext_vector_type(8)))  __bf16 v8bf;
typedef __attribute__((ext_vector_type(4)))  __bf16 v4bf;
typedef __attribute__((ext_vector_type(8)))  float  v8f;
typedef __attribute__((ext_vector_type(4)))  float  v4f;

#define BB 2
#define QQ 2048
#define KKEY 2048
#define CC 512
#define HH 8
#define DD 64
#define HD 512

// --- WMMA fragment loaders (wave32); this code's k order: k = 16*(lane>>4) + slot -----

// A matrix 16x32 bf16 from a row-major [row][k] tile: 2x ds_load_b128.
static __device__ __forceinline__ v16bf frag_a(const __bf16* p, int ld, int lane) {
  const int half = lane >> 4;
  const __bf16* base = p + (lane & 15) * ld + half * 16;
  v16bf f;
#pragma unroll
  for (int i = 0; i < 8; ++i) f[i] = base[i];
#pragma unroll
  for (int i = 0; i < 8; ++i) f[8 + i] = base[8 + i];
  return f;
}

// Fragment-major load: this lane's 16 bf16 values are contiguous (32B).
static __device__ __forceinline__ v16bf frag_ld(const __bf16* p) {
  v16bf f;
#pragma unroll
  for (int i = 0; i < 16; ++i) f[i] = p[i];
  return f;
}

// B 32x16 from a transposed tile ([n][k] row-major, e.g. K^T): 2x b128.
static __device__ __forceinline__ v16bf frag_bT(const __bf16* p, int ld, int lane) {
  const int half = lane >> 4;
  const __bf16* base = p + (lane & 15) * ld + half * 16;
  v16bf f;
#pragma unroll
  for (int i = 0; i < 16; ++i) f[i] = base[i];
  return f;
}

static __device__ __forceinline__ v8f wmma_bf16(v16bf a, v16bf b, v8f c) {
  return __builtin_amdgcn_wmma_f32_16x16x32_bf16(false, a, false, b, (short)0, c,
                                                 false, false);
}

// B-fragment LDS mapping for a 32(k) x 16(n) fragment:
//   lane = ((k>>4)<<4) | (n&15), slot = k&15

// ---------------------------------------------------------------------------
// Kernel 1: fused qkvg projections. blockIdx.z: 0=q,1=k,2=v,3=g.
// ---------------------------------------------------------------------------
__global__ void __launch_bounds__(128)
proj_kernel(const float* __restrict__ qdata, const float* __restrict__ kdata,
            const float* __restrict__ wq, const float* __restrict__ wk,
            const float* __restrict__ wv, const float* __restrict__ wg,
            const float* __restrict__ bg,
            __bf16* __restrict__ Qp, __bf16* __restrict__ Kp,
            __bf16* __restrict__ Vp, float* __restrict__ G) {
  __shared__ __align__(16) __bf16 As[2][64][40];     // row-major A tiles
  __shared__ __align__(16) __bf16 Bf[2][4][32][16];  // fragment-major B tiles

  const int which = blockIdx.z;
  const int mt = blockIdx.y, nt = blockIdx.x;
  const int tid = threadIdx.x;
  const int wave = tid >> 5, lane = tid & 31;
  const int half = lane >> 4, nn = lane & 15;

  const float* A = (which == 1 || which == 2) ? kdata : qdata;
  const float* W = (which == 0) ? wq : (which == 1) ? wk : (which == 2) ? wv : wg;

  v8f z = {};
  v8f acc[4];
#pragma unroll
  for (int t = 0; t < 4; ++t) acc[t] = z;

  v4f aC[4], wC[4];                                  // in-flight global data

  auto fetch = [&](int it) {
    const int kk = it * 32;
#pragma unroll
    for (int i = 0; i < 4; ++i) {
      int c = tid + i * 128;
      int r = c >> 3, k0 = (c & 7) * 4;
      aC[i] = *(const v4f*)&A[(size_t)(mt * 64 + r) * CC + kk + k0];
    }
#pragma unroll
    for (int i = 0; i < 4; ++i) {
      int c = tid + i * 128;
      int k = c >> 4, n0 = (c & 15) * 4;
      wC[i] = *(const v4f*)&W[(size_t)(kk + k) * HD + nt * 64 + n0];
    }
  };
  auto stage = [&](int buf) {
#pragma unroll
    for (int i = 0; i < 4; ++i) {
      int c = tid + i * 128;
      int r = c >> 3, k0 = (c & 7) * 4;
      v4bf t4;
#pragma unroll
      for (int j = 0; j < 4; ++j) t4[j] = (__bf16)aC[i][j];
      *(v4bf*)&As[buf][r][k0] = t4;
    }
#pragma unroll
    for (int i = 0; i < 4; ++i) {
      int c = tid + i * 128;
      int k = c >> 4, n0 = (c & 15) * 4;
#pragma unroll
      for (int j = 0; j < 4; ++j) {
        int n = n0 + j;
        Bf[buf][n >> 4][((k >> 4) << 4) | (n & 15)][k & 15] = (__bf16)wC[i][j];
      }
    }
  };

  const int NIT = CC / 32;
  fetch(0);
  stage(0);
  __syncthreads();
  fetch(1);                                          // tile 1 in flight

  for (int it = 0; it < NIT; ++it) {
    const int cur = it & 1;

    v16bf af = frag_a(&As[cur][wave * 16][0], 40, lane);
#pragma unroll
    for (int t = 0; t < 4; ++t)
      acc[t] = wmma_bf16(af, frag_ld(&Bf[cur][t][lane][0]), acc[t]);

    if (it + 1 < NIT) {
      stage(cur ^ 1);                                // waits loads AFTER wmma
      if (it + 2 < NIT) fetch(it + 2);               // in flight over barrier
    }
    __syncthreads();
  }

  // Epilogue: scatter to head-major [B,H,rows,D] (q scaled; g through sigmoid).
#pragma unroll
  for (int t = 0; t < 4; ++t) {
#pragma unroll
    for (int r = 0; r < 8; ++r) {
      int m = mt * 64 + wave * 16 + half * 8 + r;   // 0..4095 = b*Q + row
      int n = nt * 64 + t * 16 + nn;                // 0..511  = h*D + d
      float v = acc[t][r];
      int b = m >> 11, row = m & 2047;
      int h = n >> 6, d = n & 63;
      size_t hm = ((size_t)(b * HH + h) * QQ + row) * DD + d;
      if (which == 0)      Qp[hm] = (__bf16)(v * 0.125f);   // 1/sqrt(64)
      else if (which == 1) Kp[hm] = (__bf16)v;
      else if (which == 2) Vp[hm] = (__bf16)v;
      else {
        float g = 1.0f / (1.0f + __expf(-(v + bg[n])));
        G[(size_t)m * HD + n] = g;
      }
    }
  }
}

// ---------------------------------------------------------------------------
// Kernel 2: flash attention per (b,h). Block = 64 query rows, 4 waves.
// ---------------------------------------------------------------------------
__global__ void __launch_bounds__(128)
attn_kernel(const __bf16* __restrict__ Qp, const __bf16* __restrict__ Kp,
            const __bf16* __restrict__ Vp, const float* __restrict__ G,
            const float* __restrict__ bias1, const float* __restrict__ bias2,
            __bf16* __restrict__ Ows) {
  __shared__ __align__(16) __bf16 Kt[2][64][72];        // [key][d] for frag_bT
  __shared__ __align__(16) __bf16 Vf[2][8][32][16];     // fragment-major V
  __shared__ __align__(16) __bf16 Pf[4][2][32][16];     // per-wave P (A-layout)

  const int qt = blockIdx.x;
  const int bh = blockIdx.y;                // b*H + h
  const int b = bh >> 3, h = bh & 7;
  const int tid = threadIdx.x, wave = tid >> 5, lane = tid & 31;
  const int half = lane >> 4, nn = lane & 15;

  const size_t headQ = (size_t)bh * QQ * DD;
  const size_t headK = (size_t)bh * KKEY * DD;

  // Q fragments (this wave's 16 rows) live in registers for the whole kernel.
  const __bf16* qbase = Qp + headQ + (size_t)(qt * 64 + wave * 16) * DD;
  v16bf qf0 = frag_a(qbase, DD, lane);
  v16bf qf1 = frag_a(qbase + 32, DD, lane);

  v8f z = {};
  float mrow[8], lrow[8];
  v8f o[4];
#pragma unroll
  for (int r = 0; r < 8; ++r) { mrow[r] = -1e30f; lrow[r] = 0.0f; }
#pragma unroll
  for (int t = 0; t < 4; ++t) o[t] = z;

  const float* b1 = bias1 + (size_t)b * QQ * KKEY;
  const float* b2 = bias2 + (size_t)h * QQ * KKEY;

  v8bf kC[4], vC[4];                                 // in-flight K/V tile data

  auto fetch = [&](int kt) {
#pragma unroll
    for (int i = 0; i < 4; ++i) {
      int c = tid + i * 128;
      int key = c >> 3, d0 = (c & 7) * 8;
      size_t src = headK + (size_t)(kt * 64 + key) * DD + d0;
      kC[i] = *(const v8bf*)&Kp[src];
      vC[i] = *(const v8bf*)&Vp[src];
    }
  };
  auto stage = [&](int buf) {
#pragma unroll
    for (int i = 0; i < 4; ++i) {
      int c = tid + i * 128;
      int key = c >> 3, d0 = (c & 7) * 8;
      *(v8bf*)&Kt[buf][key][d0] = kC[i];
      int ks = key >> 5, kk2 = key & 31;
#pragma unroll
      for (int j = 0; j < 8; ++j) {
        int d = d0 + j;
        Vf[buf][ks * 4 + (d >> 4)][((kk2 >> 4) << 4) | (d & 15)][kk2 & 15] =
            vC[i][j];
      }
    }
  };

  const int NKT = KKEY / 64;
  fetch(0);
  stage(0);
  __syncthreads();
  fetch(1);                                          // tile 1 in flight

  for (int kt = 0; kt < NKT; ++kt) {
    const int cur = kt & 1;

    // Prefetch bias tile kt+1 (dominant HBM stream) while computing kt.
    if (kt + 1 < NKT) {
      int prow = qt * 64 + (tid >> 1);
      size_t boff = (size_t)prow * KKEY + (kt + 1) * 64 + (tid & 1) * 32;
      __builtin_prefetch(b1 + boff, 0, 0);
      __builtin_prefetch(b2 + boff, 0, 0);
    }

    // S = Q * K^T   (16 rows x 64 keys per wave)
    v8f s[4];
#pragma unroll
    for (int t = 0; t < 4; ++t) {
      v16bf bk0 = frag_bT(&Kt[cur][t * 16][0], 72, lane);
      v16bf bk1 = frag_bT(&Kt[cur][t * 16][32], 72, lane);
      s[t] = wmma_bf16(qf0, bk0, z);
      s[t] = wmma_bf16(qf1, bk1, s[t]);
    }

    // Additive biases, loaded directly into accumulator layout.
#pragma unroll
    for (int t = 0; t < 4; ++t) {
#pragma unroll
      for (int r = 0; r < 8; ++r) {
        int qrow = qt * 64 + wave * 16 + half * 8 + r;
        int kcol = kt * 64 + t * 16 + nn;
        size_t off = (size_t)qrow * KKEY + kcol;
        s[t][r] += b1[off] + b2[off];
      }
    }

    // Row max across 4 n-tiles + 16 lanes (xor stays within each half-wave).
    float mnew[8];
#pragma unroll
    for (int r = 0; r < 8; ++r) {
      float v = fmaxf(fmaxf(s[0][r], s[1][r]), fmaxf(s[2][r], s[3][r]));
#pragma unroll
      for (int msk = 1; msk < 16; msk <<= 1)
        v = fmaxf(v, __shfl_xor(v, msk, 32));
      mnew[r] = fmaxf(v, mrow[r]);
    }

    // P = exp(S - m); write straight into A-fragment layout (wave-private).
    float rs[8];
#pragma unroll
    for (int r = 0; r < 8; ++r) rs[r] = 0.0f;
#pragma unroll
    for (int t = 0; t < 4; ++t) {
#pragma unroll
      for (int r = 0; r < 8; ++r) {
        float p = __expf(s[t][r] - mnew[r]);
        rs[r] += p;
        // element (row = half*8+r, k = t*16+nn) of the 16x64 P block:
        Pf[wave][t >> 1][((t & 1) << 4) + half * 8 + r]
          [nn] = (__bf16)p;
      }
    }
#pragma unroll
    for (int r = 0; r < 8; ++r) {
#pragma unroll
      for (int msk = 1; msk < 16; msk <<= 1)
        rs[r] += __shfl_xor(rs[r], msk, 32);
      float sc = __expf(mrow[r] - mnew[r]);
      lrow[r] = lrow[r] * sc + rs[r];
      mrow[r] = mnew[r];
#pragma unroll
      for (int t = 0; t < 4; ++t) o[t][r] *= sc;
    }

    // O += P * V  (all fragment loads contiguous 32B -> ds_load_b128 x2)
    v16bf pf0 = frag_ld(&Pf[wave][0][lane][0]);
    v16bf pf1 = frag_ld(&Pf[wave][1][lane][0]);
#pragma unroll
    for (int t = 0; t < 4; ++t) {
      o[t] = wmma_bf16(pf0, frag_ld(&Vf[cur][t][lane][0]), o[t]);
      o[t] = wmma_bf16(pf1, frag_ld(&Vf[cur][4 + t][lane][0]), o[t]);
    }

    if (kt + 1 < NKT) {
      stage(cur ^ 1);                                // waits loads AFTER wmma
      if (kt + 2 < NKT) fetch(kt + 2);               // in flight over barrier
    }
    __syncthreads();
  }

  // Epilogue: normalize, apply sigmoid gate, store flattened [B*Q, H*D] bf16.
#pragma unroll
  for (int t = 0; t < 4; ++t) {
#pragma unroll
    for (int r = 0; r < 8; ++r) {
      int qrow = qt * 64 + wave * 16 + half * 8 + r;
      int d = t * 16 + nn;
      size_t off = (size_t)(b * QQ + qrow) * HD + h * DD + d;
      float v = o[t][r] / lrow[r];
      Ows[off] = (__bf16)(v * G[off]);
    }
  }
}

// ---------------------------------------------------------------------------
// Kernel 3: output projection  out = Ows(bf16) @ wo + bo  -> f32 [B*Q, C]
// ---------------------------------------------------------------------------
__global__ void __launch_bounds__(128)
outproj_kernel(const __bf16* __restrict__ Ows, const float* __restrict__ wo,
               const float* __restrict__ bo, float* __restrict__ out) {
  __shared__ __align__(16) __bf16 As[2][64][40];
  __shared__ __align__(16) __bf16 Bf[2][4][32][16];
  const int mt = blockIdx.y, nt = blockIdx.x;
  const int tid = threadIdx.x, wave = tid >> 5, lane = tid & 31;
  const int half = lane >> 4, nn = lane & 15;

  v8f z = {};
  v8f acc[4];
#pragma unroll
  for (int t = 0; t < 4; ++t) acc[t] = z;

  v8bf aC[2];
  v4f wC[4];

  auto fetch = [&](int it) {
    const int kk = it * 32;
#pragma unroll
    for (int i = 0; i < 2; ++i) {
      int c = tid + i * 128;
      int r = c >> 2, k0 = (c & 3) * 8;
      aC[i] = *(const v8bf*)&Ows[(size_t)(mt * 64 + r) * HD + kk + k0];
    }
#pragma unroll
    for (int i = 0; i < 4; ++i) {
      int c = tid + i * 128;
      int k = c >> 4, n0 = (c & 15) * 4;
      wC[i] = *(const v4f*)&wo[(size_t)(kk + k) * CC + nt * 64 + n0];
    }
  };
  auto stage = [&](int buf) {
#pragma unroll
    for (int i = 0; i < 2; ++i) {
      int c = tid + i * 128;
      int r = c >> 2, k0 = (c & 3) * 8;
      *(v8bf*)&As[buf][r][k0] = aC[i];
    }
#pragma unroll
    for (int i = 0; i < 4; ++i) {
      int c = tid + i * 128;
      int k = c >> 4, n0 = (c & 15) * 4;
#pragma unroll
      for (int j = 0; j < 4; ++j) {
        int n = n0 + j;
        Bf[buf][n >> 4][((k >> 4) << 4) | (n & 15)][k & 15] = (__bf16)wC[i][j];
      }
    }
  };

  const int NIT = HD / 32;
  fetch(0);
  stage(0);
  __syncthreads();
  fetch(1);

  for (int it = 0; it < NIT; ++it) {
    const int cur = it & 1;

    v16bf af = frag_a(&As[cur][wave * 16][0], 40, lane);
#pragma unroll
    for (int t = 0; t < 4; ++t)
      acc[t] = wmma_bf16(af, frag_ld(&Bf[cur][t][lane][0]), acc[t]);

    if (it + 1 < NIT) {
      stage(cur ^ 1);
      if (it + 2 < NIT) fetch(it + 2);
    }
    __syncthreads();
  }

#pragma unroll
  for (int t = 0; t < 4; ++t) {
#pragma unroll
    for (int r = 0; r < 8; ++r) {
      int m = mt * 64 + wave * 16 + half * 8 + r;
      int n = nt * 64 + t * 16 + nn;
      out[(size_t)m * CC + n] = acc[t][r] + bo[n];
    }
  }
}

// ---------------------------------------------------------------------------
extern "C" void kernel_launch(void* const* d_in, const int* in_sizes, int n_in,
                              void* d_out, int out_size, void* d_ws, size_t ws_size,
                              hipStream_t stream) {
  const float* qdata = (const float*)d_in[0];
  const float* kdata = (const float*)d_in[1];
  const float* bias1 = (const float*)d_in[2];
  const float* bias2 = (const float*)d_in[3];
  const float* wq    = (const float*)d_in[4];
  const float* wk    = (const float*)d_in[5];
  const float* wv    = (const float*)d_in[6];
  const float* wo    = (const float*)d_in[7];
  const float* bo    = (const float*)d_in[8];
  const float* wg    = (const float*)d_in[9];
  const float* bg    = (const float*)d_in[10];
  float* out = (float*)d_out;

  char* ws = (char*)d_ws;
  __bf16* Qp  = (__bf16*)(ws);
  __bf16* Kp  = (__bf16*)(ws + (size_t)4 * 1024 * 1024);
  __bf16* Vp  = (__bf16*)(ws + (size_t)8 * 1024 * 1024);
  float*  G   = (float*) (ws + (size_t)12 * 1024 * 1024);
  __bf16* Ows = (__bf16*)(ws + (size_t)20 * 1024 * 1024);

  dim3 g1(HD / 64, (BB * QQ) / 64, 4);            // 8 x 64 x 4
  proj_kernel<<<g1, 128, 0, stream>>>(qdata, kdata, wq, wk, wv, wg, bg,
                                      Qp, Kp, Vp, G);

  dim3 g2(QQ / 64, BB * HH);                      // 32 x 16
  attn_kernel<<<g2, 128, 0, stream>>>(Qp, Kp, Vp, G, bias1, bias2, Ows);

  dim3 g3(CC / 64, (BB * QQ) / 64);               // 8 x 64
  outproj_kernel<<<g3, 128, 0, stream>>>(Ows, wo, bo, out);
}
